// WaveFlow_38551626449094
// MI455X (gfx1250) — compile-verified, matrix operand order checked by reading, not run on hardware
//
#include <hip/hip_runtime.h>
#include <hip/hip_bf16.h>

// ---------------------------------------------------------------------------
// Types for CDNA5 WMMA bf16 path
// ---------------------------------------------------------------------------
typedef __bf16 bf16_t;
typedef __attribute__((ext_vector_type(16))) __bf16 v16bf;
typedef __attribute__((ext_vector_type(8)))  __bf16 v8bf;
typedef __attribute__((ext_vector_type(8)))  float  v8f;

#define N_FLOW   8
#define N_LAYER  8
#define RES      128
#define CIN      80
#define CINP     96          // CIN padded to multiple of 32 for K blocks
#define HROWS    16
#define WCOLS    2048
#define RROWS    15          // 16-1 rows used by wavenet
#define TLEN     32768
#define HALO     128         // max dw = 2^7
#define HBW      (WCOLS + 2*HALO)   // 2304 padded columns for h (bf16)
#define SQRT_HALF 0.70710678118654752f

// ---------------------------------------------------------------------------
// Workspace layout (byte offsets; all multiples of 256)
// ---------------------------------------------------------------------------
static constexpr size_t OFF_CMID   = 0;                                      // 80*2048*4
static constexpr size_t OFF_CUP    = OFF_CMID   + (size_t)80*2048*4;         // 80*32768*4
static constexpr size_t OFF_C2P    = OFF_CUP    + (size_t)80*32768*4;        // 2*16*2048*96*2
static constexpr size_t OFF_WG     = OFF_C2P    + (size_t)2*16*2048*96*2;    // 64*256*9*128*2
static constexpr size_t OFF_WC     = OFF_WG     + (size_t)64*256*9*128*2;    // 64*256*96*2
static constexpr size_t OFF_WR     = OFF_WC     + (size_t)64*256*96*2;       // 64*128*128*2
static constexpr size_t OFF_HF     = OFF_WR     + (size_t)64*128*128*2;      // 15*2048*128*4
static constexpr size_t OFF_HB     = OFF_HF     + (size_t)RROWS*WCOLS*RES*4; // 15*2304*128*2 (halo)
static constexpr size_t OFF_GB     = OFF_HB     + (size_t)RROWS*HBW*RES*2;   // 15*2048*128*2
static constexpr size_t OFF_SKIP   = OFF_GB     + (size_t)RROWS*WCOLS*RES*2; // f32
static constexpr size_t OFF_OUTA   = OFF_SKIP   + (size_t)RROWS*WCOLS*RES*4;
static constexpr size_t OFF_OUTB   = OFF_OUTA   + (size_t)HROWS*WCOLS*4;
static constexpr size_t OFF_LOGDET = OFF_OUTB   + (size_t)HROWS*WCOLS*4;

// ---------------------------------------------------------------------------
// WMMA helpers
// ---------------------------------------------------------------------------
__device__ __forceinline__ v8f wmma_bf16(v16bf a, v16bf b, v8f c) {
  return __builtin_amdgcn_wmma_f32_16x16x32_bf16(
      /*neg_a=*/false, a, /*neg_b=*/false, b,
      /*c_mod=*/(short)0, c, /*reuse_a=*/false, /*reuse_b=*/false);
}

// Load a lane's A/B fragment: 8 bf16 at p plus 8 bf16 at p+16 (two b128 loads).
__device__ __forceinline__ v16bf load_frag(const bf16_t* p) {
  v8bf lo = *(const v8bf*)p;
  v8bf hi = *(const v8bf*)(p + 16);
  v16bf r;
#pragma unroll
  for (int i = 0; i < 8; ++i) { r[i] = lo[i]; r[i + 8] = hi[i]; }
  return r;
}

// ---------------------------------------------------------------------------
// Weight conversion kernels (fp32 -> bf16, GEMM-friendly layouts)
// ---------------------------------------------------------------------------
// fg_w (8,8,256,128,3,3) -> Wg [fl][oc][tap][ic]  (ic contiguous)
__global__ void conv_wg_kernel(const float* __restrict__ fg, bf16_t* __restrict__ wg) {
  size_t o = (size_t)blockIdx.x * blockDim.x + threadIdx.x;
  if (o >= (size_t)64 * 256 * 9 * 128) return;
  int ic  = (int)(o & 127);
  size_t t = o >> 7;
  int tap = (int)(t % 9); t /= 9;
  int oc  = (int)(t % 256);
  int fl  = (int)(t / 256);
  int kh = tap / 3, kw = tap % 3;
  size_t src = ((((size_t)fl * 256 + oc) * 128 + ic) * 3 + kh) * 3 + kw;
  wg[o] = (bf16_t)fg[src];
}

// fgc_w (8,8,256,80) -> Wc [fl][oc][icp96] zero-padded
__global__ void conv_wc_kernel(const float* __restrict__ fgc, bf16_t* __restrict__ wc) {
  size_t o = (size_t)blockIdx.x * blockDim.x + threadIdx.x;
  if (o >= (size_t)64 * 256 * CINP) return;
  int ic  = (int)(o % CINP);
  size_t t = o / CINP;
  int oc  = (int)(t % 256);
  int fl  = (int)(t / 256);
  float v = (ic < CIN) ? fgc[((size_t)fl * 256 + oc) * CIN + ic] : 0.0f;
  wc[o] = (bf16_t)v;
}

// rs_w (8,8,128,128) -> bf16, same layout
__global__ void conv_wr_kernel(const float* __restrict__ rs, bf16_t* __restrict__ wr) {
  size_t o = (size_t)blockIdx.x * blockDim.x + threadIdx.x;
  if (o >= (size_t)64 * 128 * 128) return;
  wr[o] = (bf16_t)rs[o];
}

// ---------------------------------------------------------------------------
// Upsampler: transposed conv (3x32 kernel, W lhs_dilation 16, pad H(1,1) W(23,23))
// ---------------------------------------------------------------------------
__global__ void upsample_kernel(const float* __restrict__ in, int Tin,
                                float* __restrict__ outp, int Tout,
                                const float* __restrict__ w, const float* __restrict__ b) {
  size_t i = (size_t)blockIdx.x * blockDim.x + threadIdx.x;
  if (i >= (size_t)80 * Tout) return;
  int m  = (int)(i / Tout);
  int wo = (int)(i % Tout);
  int j0 = ((23 - wo) % 16 + 16) % 16;
  float acc = 0.0f;
#pragma unroll
  for (int s = 0; s < 2; ++s) {
    int kw = j0 + s * 16;
    int t  = wo + kw - 23;
    if (t < 0) continue;
    int u = t >> 4;                  // t is a multiple of 16 by construction
    if (u >= Tin) continue;
#pragma unroll
    for (int kh = 0; kh < 3; ++kh) {
      int mm = m + kh - 1;
      if (mm < 0 || mm >= 80) continue;
      acc += w[(2 - kh) * 32 + (31 - kw)] * in[(size_t)mm * Tin + u];
    }
  }
  acc += b[0];
  outp[i] = (acc > 0.0f) ? acc : 0.4f * acc;   // leaky(0.4)
}

// Build c2 in bf16, padded to 96 ch, both H orientations: [orient][row][w][icp]
__global__ void c2prep_kernel(const float* __restrict__ cup, bf16_t* __restrict__ c2p) {
  size_t i = (size_t)blockIdx.x * blockDim.x + threadIdx.x;
  if (i >= (size_t)2 * HROWS * WCOLS * CINP) return;
  int ic  = (int)(i % CINP);
  size_t t = i / CINP;
  int w   = (int)(t % WCOLS); t /= WCOLS;
  int row = (int)(t % HROWS);
  int orient = (int)(t / HROWS);
  int rr = orient ? (HROWS - 1 - row) : row;
  float v = (ic < CIN) ? cup[(size_t)ic * TLEN + (size_t)w * HROWS + rr] : 0.0f;
  c2p[i] = (bf16_t)v;
}

// squeeze2d init of out buffer + zero logdet
__global__ void init_out_kernel(const float* __restrict__ x, float* __restrict__ outA,
                                float* __restrict__ logdet) {
  int i = blockIdx.x * blockDim.x + threadIdx.x;
  if (i < HROWS * WCOLS) {
    int h = i >> 11, w = i & (WCOLS - 1);
    outA[i] = x[(size_t)w * HROWS + h];
  }
  if (i == 0) logdet[0] = 0.0f;
}

// ---------------------------------------------------------------------------
// Front 1x1 conv (128 x 1); writes h f32 (interior) and halo-padded bf16;
// zeroes the halo columns and skip buffer.
// ---------------------------------------------------------------------------
__global__ void front_kernel(const float* __restrict__ out_cur,
                             const float* __restrict__ fw, const float* __restrict__ fb,
                             float* __restrict__ hf, bf16_t* __restrict__ hb,
                             float* __restrict__ skip) {
  size_t i = (size_t)blockIdx.x * blockDim.x + threadIdx.x;
  if (i >= (size_t)RROWS * HBW * RES) return;
  int ic = (int)(i & (RES - 1));
  size_t pos = i >> 7;                     // r*HBW + col
  int col = (int)(pos % HBW);
  int r   = (int)(pos / HBW);
  if (col < HALO || col >= WCOLS + HALO) { // zero halo
    hb[i] = (bf16_t)0.0f;
    return;
  }
  int w = col - HALO;
  float v = fw[ic] * out_cur[(size_t)r * WCOLS + w] + fb[ic];
  hb[i] = (bf16_t)v;
  size_t ii = ((size_t)r * WCOLS + w) * RES + ic;
  hf[ii]   = v;
  skip[ii] = 0.0f;
}

// ---------------------------------------------------------------------------
// Fused dilated-conv + cond + gate kernel (WMMA bf16).
// Block = 256 thr = 8 waves. Tile: M=256 (all out ch), N=64, fixed row r.
// Wave wv owns M-tiles {wv*16 (tanh), wv*16+128 (sigmoid)} x 4 N-tiles.
// K = 9 taps * 128 ic + 96 cond ic -> 39 WMMA K-steps, 8 WMMA per step.
// Branch-free inner loop thanks to the zero halo in hb.
// ---------------------------------------------------------------------------
__global__ void __launch_bounds__(256)
gate_kernel(const bf16_t* __restrict__ Wg,    // [256][9][128] for this (f,l)
            const bf16_t* __restrict__ Wc,    // [256][96]
            const bf16_t* __restrict__ hb,    // [15][HBW][128], halo zeroed
            const bf16_t* __restrict__ c2p,   // [16][2048][96] (orientation chosen)
            const float*  __restrict__ fg_b,  // [256]
            const float*  __restrict__ fgc_b, // [256]
            float*  __restrict__ skip,        // [15][2048][128]
            bf16_t* __restrict__ gb,          // [15][2048][128]
            int dh, int dw) {
  const int blk  = blockIdx.x;
  const int r    = blk >> 5;               // 32 wblocks of 64 per row
  const int w0   = (blk & 31) * 64;
  const int lane = threadIdx.x & 31;
  const int wv   = threadIdx.x >> 5;
  const int m0   = wv * 16;
  const int hi   = lane >> 4;
  const int ln   = lane & 15;

  v8f acc[8];                              // [half(t=0..3, s=4..7)]
#pragma unroll
  for (int a = 0; a < 8; ++a)
#pragma unroll
    for (int j = 0; j < 8; ++j) acc[a][j] = 0.0f;

  const bf16_t* wg_t = Wg + (size_t)(m0 + ln) * (9 * 128) + hi * 8;
  const bf16_t* wg_s = Wg + (size_t)(m0 + 128 + ln) * (9 * 128) + hi * 8;

  // ---- dilated causal conv: 9 taps x 4 ic-blocks, branch-free B loads ----
  for (int tap = 0; tap < 9; ++tap) {
    const int kh = tap / 3, kw = tap % 3;
    const int rr = r + dh * (kh - 2);      // causal rows: r-2dh, r-dh, r
    if (rr < 0) continue;
    const bf16_t* hrow = hb + (size_t)rr * HBW * RES + hi * 8;
    const int wbase = w0 + ln + dw * (kw - 1) + HALO;  // always in [0, HBW)
#pragma unroll
    for (int kb = 0; kb < 4; ++kb) {
      const int k0 = kb * 32;
      const v16bf a_t = load_frag(wg_t + tap * 128 + k0);
      const v16bf a_s = load_frag(wg_s + tap * 128 + k0);
#pragma unroll
      for (int nt = 0; nt < 4; ++nt) {
        const v16bf b = load_frag(hrow + (size_t)(wbase + nt * 16) * RES + k0);
        acc[nt]     = wmma_bf16(a_t, b, acc[nt]);
        acc[4 + nt] = wmma_bf16(a_s, b, acc[4 + nt]);
      }
    }
  }

  // ---- conditioning 1x1 (96 padded ch, row r+1) ----
  {
    const bf16_t* crow = c2p + (size_t)(r + 1) * WCOLS * CINP + hi * 8;
    const bf16_t* wc_t = Wc + (size_t)(m0 + ln) * CINP + hi * 8;
    const bf16_t* wc_s = Wc + (size_t)(m0 + 128 + ln) * CINP + hi * 8;
#pragma unroll
    for (int kb = 0; kb < 3; ++kb) {
      const int k0 = kb * 32;
      const v16bf a_t = load_frag(wc_t + k0);
      const v16bf a_s = load_frag(wc_s + k0);
#pragma unroll
      for (int nt = 0; nt < 4; ++nt) {
        const v16bf b = load_frag(crow + (size_t)(w0 + nt * 16 + ln) * CINP + k0);
        acc[nt]     = wmma_bf16(a_t, b, acc[nt]);
        acc[4 + nt] = wmma_bf16(a_s, b, acc[4 + nt]);
      }
    }
  }

  // ---- epilogue: bias (hoisted), gate, skip += g, g -> bf16 ----
  float bt[8], bs[8];
#pragma unroll
  for (int v = 0; v < 8; ++v) {
    const int m = m0 + v + hi * 8;
    bt[v] = fg_b[m] + fgc_b[m];
    bs[v] = fg_b[m + 128] + fgc_b[m + 128];
  }
#pragma unroll
  for (int nt = 0; nt < 4; ++nt) {
    const int w = w0 + nt * 16 + ln;
#pragma unroll
    for (int v = 0; v < 8; ++v) {
      const int m = v + hi * 8;            // C/D layout: lane<16 M=v, lane>=16 M=v+8
      const float tv = tanhf(acc[nt][v] + bt[v]);
      const float sv = 1.0f / (1.0f + expf(-(acc[4 + nt][v] + bs[v])));
      const float g  = tv * sv;
      const size_t idx = ((size_t)r * WCOLS + w) * RES + (m0 + m);
      skip[idx] += g;
      gb[idx] = (bf16_t)g;
    }
  }
}

// ---------------------------------------------------------------------------
// Residual 1x1 (128x128) WMMA GEMM: h = (h + rs_w@g + rs_b) * sqrt(1/2)
// Block = 256 thr = 8 waves; wave wv owns M-tile wv*16, 4 N-tiles.
// ---------------------------------------------------------------------------
__global__ void __launch_bounds__(256)
resid_kernel(const bf16_t* __restrict__ Wr,   // [128][128]
             const float*  __restrict__ rs_b, // [128]
             const bf16_t* __restrict__ gb,   // [15][2048][128]
             float*  __restrict__ hf,         // [15][2048][128]
             bf16_t* __restrict__ hb) {       // [15][HBW][128]
  const int blk  = blockIdx.x;
  const int r    = blk >> 5;
  const int w0   = (blk & 31) * 64;
  const int lane = threadIdx.x & 31;
  const int wv   = threadIdx.x >> 5;
  const int m0   = wv * 16;
  const int hi   = lane >> 4;
  const int ln   = lane & 15;

  v8f acc[4];
#pragma unroll
  for (int a = 0; a < 4; ++a)
#pragma unroll
    for (int j = 0; j < 8; ++j) acc[a][j] = 0.0f;

  const bf16_t* wr_p = Wr + (size_t)(m0 + ln) * 128 + hi * 8;
  const bf16_t* grow = gb + ((size_t)r * WCOLS) * RES + hi * 8;
#pragma unroll
  for (int kb = 0; kb < 4; ++kb) {
    const int k0 = kb * 32;
    const v16bf a = load_frag(wr_p + k0);
#pragma unroll
    for (int nt = 0; nt < 4; ++nt) {
      const v16bf b = load_frag(grow + (size_t)(w0 + nt * 16 + ln) * RES + k0);
      acc[nt] = wmma_bf16(a, b, acc[nt]);
    }
  }

  float rb[8];
#pragma unroll
  for (int v = 0; v < 8; ++v) rb[v] = rs_b[m0 + v + hi * 8];

#pragma unroll
  for (int nt = 0; nt < 4; ++nt) {
    const int w = w0 + nt * 16 + ln;
#pragma unroll
    for (int v = 0; v < 8; ++v) {
      const int m = v + hi * 8;
      const size_t idx = ((size_t)r * WCOLS + w) * RES + (m0 + m);
      const float nv = (hf[idx] + acc[nt][v] + rb[v]) * SQRT_HALF;
      hf[idx] = nv;
      hb[((size_t)r * HBW + (w + HALO)) * RES + (m0 + m)] = (bf16_t)nv;
    }
  }
}

// ---------------------------------------------------------------------------
// Projection + affine coupling + H-reversal (fused) + logdet reduction
// ---------------------------------------------------------------------------
__global__ void __launch_bounds__(256)
couple_kernel(const float* __restrict__ skip,  // [15][2048][128]
              const float* __restrict__ pw,    // [2][128]
              const float* __restrict__ pb,    // [2]
              const float* __restrict__ ps,    // [2]
              const float* __restrict__ out_cur,
              float* __restrict__ out_next,
              float* __restrict__ logdet) {
  const int idx = blockIdx.x * blockDim.x + threadIdx.x;
  float ld = 0.0f;
  if (idx < RROWS * WCOLS) {
    const int r = idx >> 11, w = idx & (WCOLS - 1);
    const float* sp = skip + (size_t)idx * RES;
    float s0 = 0.0f, s1 = 0.0f;
#pragma unroll 4
    for (int ic = 0; ic < RES; ++ic) {
      const float v = sp[ic];
      s0 += pw[ic] * v;
      s1 += pw[RES + ic] * v;
    }
    const float log_s = (s0 + pb[0]) * expf(3.0f * ps[0]);
    const float tval  = (s1 + pb[1]) * expf(3.0f * ps[1]);
    const float xn = out_cur[(size_t)(r + 1) * WCOLS + w] * expf(log_s) + tval;
    out_next[(size_t)(14 - r) * WCOLS + w] = xn;          // reversed write: 15-(r+1)
    if (r == 0) out_next[(size_t)15 * WCOLS + w] = out_cur[w];  // row0 -> reversed row15
    ld = log_s;
  }
  __shared__ float red[256];
  red[threadIdx.x] = ld;
  __syncthreads();
  for (int s = 128; s > 0; s >>= 1) {
    if (threadIdx.x < s) red[threadIdx.x] += red[threadIdx.x + s];
    __syncthreads();
  }
  if (threadIdx.x == 0) atomicAdd(logdet, red[0]);
}

__global__ void finalize_kernel(const float* __restrict__ outF,
                                const float* __restrict__ logdet,
                                float* __restrict__ d_out) {
  const int i = blockIdx.x * blockDim.x + threadIdx.x;
  if (i < HROWS * WCOLS) d_out[i] = outF[i];
  if (i == HROWS * WCOLS) d_out[i] = logdet[0];
}

// ---------------------------------------------------------------------------
// Host driver
// ---------------------------------------------------------------------------
extern "C" void kernel_launch(void* const* d_in, const int* in_sizes, int n_in,
                              void* d_out, int out_size, void* d_ws, size_t ws_size,
                              hipStream_t stream) {
  (void)in_sizes; (void)n_in; (void)out_size; (void)ws_size;
  const float* x       = (const float*)d_in[0];
  const float* c       = (const float*)d_in[1];
  const float* front_w = (const float*)d_in[2];
  const float* front_b = (const float*)d_in[3];
  const float* fg_w    = (const float*)d_in[4];
  const float* fg_b    = (const float*)d_in[5];
  const float* fgc_w   = (const float*)d_in[6];
  const float* fgc_b   = (const float*)d_in[7];
  const float* rs_w    = (const float*)d_in[8];
  const float* rs_b    = (const float*)d_in[9];
  const float* proj_w  = (const float*)d_in[10];
  const float* proj_b  = (const float*)d_in[11];
  const float* proj_s  = (const float*)d_in[12];
  const float* up_w1   = (const float*)d_in[13];
  const float* up_b1   = (const float*)d_in[14];
  const float* up_w2   = (const float*)d_in[15];
  const float* up_b2   = (const float*)d_in[16];

  char* ws = (char*)d_ws;
  float*  cmid  = (float*)(ws + OFF_CMID);
  float*  cup   = (float*)(ws + OFF_CUP);
  bf16_t* c2p   = (bf16_t*)(ws + OFF_C2P);
  bf16_t* Wg    = (bf16_t*)(ws + OFF_WG);
  bf16_t* Wc    = (bf16_t*)(ws + OFF_WC);
  bf16_t* Wr    = (bf16_t*)(ws + OFF_WR);
  float*  hf    = (float*)(ws + OFF_HF);
  bf16_t* hb    = (bf16_t*)(ws + OFF_HB);
  bf16_t* gb    = (bf16_t*)(ws + OFF_GB);
  float*  skip  = (float*)(ws + OFF_SKIP);
  float*  outA  = (float*)(ws + OFF_OUTA);
  float*  outB  = (float*)(ws + OFF_OUTB);
  float*  ldet  = (float*)(ws + OFF_LOGDET);

  const int TB = 256;
  auto blocks = [](size_t n, int tb) { return (unsigned)((n + tb - 1) / tb); };

  // 1) weight conversion to bf16
  conv_wg_kernel<<<blocks((size_t)64*256*9*128, TB), TB, 0, stream>>>(fg_w, Wg);
  conv_wc_kernel<<<blocks((size_t)64*256*CINP, TB), TB, 0, stream>>>(fgc_w, Wc);
  conv_wr_kernel<<<blocks((size_t)64*128*128, TB), TB, 0, stream>>>(rs_w, Wr);

  // 2) mel upsampling (x16 twice) + leaky
  upsample_kernel<<<blocks((size_t)80*2048, TB), TB, 0, stream>>>(c, 128, cmid, 2048, up_w1, up_b1);
  upsample_kernel<<<blocks((size_t)80*TLEN, TB), TB, 0, stream>>>(cmid, 2048, cup, TLEN, up_w2, up_b2);

  // 3) squeeze + bf16-pad c2, both orientations
  c2prep_kernel<<<blocks((size_t)2*HROWS*WCOLS*CINP, TB), TB, 0, stream>>>(cup, c2p);

  // 4) squeeze x into outA; zero logdet
  init_out_kernel<<<blocks((size_t)HROWS*WCOLS, TB), TB, 0, stream>>>(x, outA, ldet);

  const unsigned gemm_blocks = RROWS * (WCOLS / 64);   // 480

  // 5) flows
  for (int f = 0; f < N_FLOW; ++f) {
    float* cur = (f & 1) ? outB : outA;
    float* nxt = (f & 1) ? outA : outB;
    const bf16_t* c2o = c2p + (size_t)(f & 1) * HROWS * WCOLS * CINP;

    front_kernel<<<blocks((size_t)RROWS*HBW*RES, TB), TB, 0, stream>>>(
        cur, front_w + (size_t)f * RES, front_b + (size_t)f * RES, hf, hb, skip);

    for (int l = 0; l < N_LAYER; ++l) {
      const int fl = f * N_LAYER + l;
      const int dh = 1 << (l % 3);
      const int dw = 1 << l;
      gate_kernel<<<gemm_blocks, TB, 0, stream>>>(
          Wg + (size_t)fl * 256 * 9 * 128,
          Wc + (size_t)fl * 256 * CINP,
          hb, c2o,
          fg_b + (size_t)fl * 256, fgc_b + (size_t)fl * 256,
          skip, gb, dh, dw);
      if (l < N_LAYER - 1) {
        resid_kernel<<<gemm_blocks, TB, 0, stream>>>(
            Wr + (size_t)fl * 128 * 128, rs_b + (size_t)fl * 128, gb, hf, hb);
      }
    }

    couple_kernel<<<blocks((size_t)RROWS*WCOLS, TB), TB, 0, stream>>>(
        skip, proj_w + (size_t)f * 2 * RES, proj_b + (size_t)f * 2,
        proj_s + (size_t)f * 2, cur, nxt, ldet);
  }

  // after 8 flows the result lives in outA (f=7 wrote nxt=outA)
  finalize_kernel<<<blocks((size_t)HROWS*WCOLS + 1, TB), TB, 0, stream>>>(outA, ldet, (float*)d_out);
}
